// Model_55224689492143
// MI455X (gfx1250) — matrix operand order repaired and checked by reading, not compiled, not run on hardware
//
#include <hip/hip_runtime.h>
#include <hip/hip_fp16.h>
#include <math.h>

typedef __attribute__((ext_vector_type(16))) _Float16 v16h;
typedef __attribute__((ext_vector_type(8)))  float    v8f;

// Problem constants (from reference): B=8, N=512, D=64, O=64
#define NB   8
#define NN   512
#define ND   64
#define NO   64

// Branch-free tanh: prefer the CDNA5 hardware v_tanh_f32; otherwise a
// divergence-free formulation from v_exp_f32 + v_rcp_f32.
__device__ __forceinline__ float fast_tanhf(float v) {
#if __has_builtin(__builtin_amdgcn_tanhf)
    return __builtin_amdgcn_tanhf(v);
#elif __has_builtin(__builtin_amdgcn_tanh_f32)
    return __builtin_amdgcn_tanh_f32(v);
#else
    // tanh(x) = 1 - 2/(exp(2x)+1); exp(+inf)->inf gives 1, exp(-inf)->0 gives -1
    return 1.0f - 2.0f / (__expf(2.0f * v) + 1.0f);
#endif
}

__global__ __launch_bounds__(256)
void gat_fused_kernel(const float* __restrict__ x,
                      const float* __restrict__ W_ap,
                      const float* __restrict__ b_ap,
                      const float* __restrict__ att_w,
                      const float* __restrict__ W_pa,
                      const float* __restrict__ b_pa,
                      const float* __restrict__ W_po,
                      const float* __restrict__ b_po,
                      const float* __restrict__ bn_gamma,
                      const float* __restrict__ bn_beta,
                      const float* __restrict__ run_mean,
                      const float* __restrict__ run_var,
                      float* __restrict__ out)
{
    const int tid  = threadIdx.x;
    const int wave = tid >> 5;
    const int lane = tid & 31;
    const int lg   = lane & 15;   // lane within half-group
    const int hi   = lane >> 4;   // which half-group

    const int bi = blockIdx.x;    // b*512 + i
    const int b  = bi >> 9;
    const int i  = bi & 511;

    __shared__ _Float16 sM[ND * NO];   // M_i = diag(x_i) @ W_ap, f16, [d][o]
    __shared__ float    sXi[ND];       // x[b,i,:]
    __shared__ float    sAux[2 * NO];  // [0..63]=att_w, [64..127]=b_ap
    __shared__ float    sScores[NN];   // scores -> att_map
    __shared__ float    sRed[256];     // reductions
    __shared__ float    sPart[4][ND];  // agg partials
    __shared__ float    sAgg[ND];      // agg vector

    const float* xb = x + (size_t)b * NN * ND;

    // ---- stage x_i, att_w, b_ap ----
    if (tid < ND) {
        sXi[tid]        = xb[(size_t)i * ND + tid];
        sAux[tid]       = att_w[tid];        // shape (64,1) flat
        sAux[NO + tid]  = b_ap[tid];
    }
    __syncthreads();

    // ---- M_i[d][o] = x_i[d] * W_ap[d][o], as f16 ----
    for (int idx = tid; idx < ND * NO; idx += 256) {
        int d = idx >> 6;
        sM[idx] = (_Float16)(sXi[d] * W_ap[idx]);
    }
    __syncthreads();

    // ---- preload B fragments (32x16 f16, contiguous k = 16*hi + s, same order as A), reused all j-tiles ----
    // B layout: VGPR v holds K = 2v + (hi?16:0) and K+1; column N = lg.
    v16h bf[8];   // [ot*2 + kt]
#pragma unroll
    for (int ot = 0; ot < 4; ++ot) {
#pragma unroll
        for (int kt = 0; kt < 2; ++kt) {
            v16h f;
#pragma unroll
            for (int v = 0; v < 8; ++v) {
                int k = kt * 32 + 2 * v + (hi ? 16 : 0);
                int o = ot * 16 + lg;
                f[2 * v]     = sM[k * NO + o];
                f[2 * v + 1] = sM[(k + 1) * NO + o];
            }
            bf[ot * 2 + kt] = f;
        }
    }

    // ---- main loop: att tile = X_b[jt] @ M_i ; tanh ; dot with att_w -> scores ----
    for (int jt = wave; jt < NN / 16; jt += 8) {
        // A fragments (16x32 f16): lane lg = row M; VGPR v: K = 2v + (hi?16:0)
        v16h a0, a1;
        const float* xr = xb + (size_t)(jt * 16 + lg) * ND;
#pragma unroll
        for (int v = 0; v < 8; ++v) {
            int kb = 2 * v + (hi ? 16 : 0);
            float2 t0 = *(const float2*)(xr + kb);        // kt = 0
            float2 t1 = *(const float2*)(xr + 32 + kb);   // kt = 1
            a0[2 * v]     = (_Float16)t0.x;
            a0[2 * v + 1] = (_Float16)t0.y;
            a1[2 * v]     = (_Float16)t1.x;
            a1[2 * v + 1] = (_Float16)t1.y;
        }

        float sj[8];
#pragma unroll
        for (int v = 0; v < 8; ++v) sj[v] = 0.f;

#pragma unroll
        for (int ot = 0; ot < 4; ++ot) {
            float bias = sAux[NO + ot * 16 + lg];   // b_ap[o], same for all 8 rows of this lane
            float aw   = sAux[ot * 16 + lg];        // att_w[o]
            v8f acc;
#pragma unroll
            for (int v = 0; v < 8; ++v) acc[v] = bias;
            // D = A*B + C ; f32 accumulate
            acc = __builtin_amdgcn_wmma_f32_16x16x32_f16(
                      false, a0, false, bf[ot * 2 + 0], (short)0, acc, false, false);
            acc = __builtin_amdgcn_wmma_f32_16x16x32_f16(
                      false, a1, false, bf[ot * 2 + 1], (short)0, acc, false, false);
#pragma unroll
            for (int v = 0; v < 8; ++v) sj[v] += fast_tanhf(acc[v]) * aw;
        }

        // reduce over the 16 N-lanes of each half-group (butterfly stays within halves)
#pragma unroll
        for (int m = 1; m < 16; m <<= 1) {
#pragma unroll
            for (int v = 0; v < 8; ++v) sj[v] += __shfl_xor(sj[v], m, 32);
        }
        if (lg == 0) {
#pragma unroll
            for (int v = 0; v < 8; ++v)
                sScores[jt * 16 + hi * 8 + v] = sj[v];   // row M = v + 8*hi
        }
    }
    __syncthreads();

    // ---- softmax over j (512 values) ----
    float m = -1e30f;
    for (int j = tid; j < NN; j += 256) m = fmaxf(m, sScores[j]);
    sRed[tid] = m;
    __syncthreads();
    for (int s = 128; s > 0; s >>= 1) {
        if (tid < s) sRed[tid] = fmaxf(sRed[tid], sRed[tid + s]);
        __syncthreads();
    }
    const float mx = sRed[0];
    __syncthreads();

    float lsum = 0.f;
    for (int j = tid; j < NN; j += 256) {
        float e = __expf(sScores[j] - mx);
        sScores[j] = e;
        lsum += e;
    }
    sRed[tid] = lsum;
    __syncthreads();
    for (int s = 128; s > 0; s >>= 1) {
        if (tid < s) sRed[tid] += sRed[tid + s];
        __syncthreads();
    }
    const float inv = 1.0f / sRed[0];
    __syncthreads();
    for (int j = tid; j < NN; j += 256) sScores[j] *= inv;
    __syncthreads();

    // ---- agg[d] = sum_j att_map[j] * x[b,j,d]  (partials over 4 j-chunks) ----
    {
        const int c = tid >> 6;       // j-chunk 0..3
        const int d = tid & 63;
        const float* xc = xb + (size_t)c * 128 * ND + d;
        float p = 0.f;
        for (int j = 0; j < 128; ++j) p += sScores[c * 128 + j] * xc[(size_t)j * ND];
        sPart[c][d] = p;
    }
    __syncthreads();
    if (tid < ND)
        sAgg[tid] = sPart[0][tid] + sPart[1][tid] + sPart[2][tid] + sPart[3][tid];
    __syncthreads();

    // ---- h = agg@W_pa + b_pa + x_i@W_po + b_po ; BN(eval) ; SELU ----
    if (tid < NO) {
        const int o = tid;
        float acc = b_pa[o] + b_po[o];
        for (int d = 0; d < ND; ++d)
            acc += sAgg[d] * W_pa[d * NO + o] + sXi[d] * W_po[d * NO + o];
        float h = (acc - run_mean[o]) * rsqrtf(run_var[o] + 1e-5f) * bn_gamma[o] + bn_beta[o];
        const float SELU_L = 1.0507009873554805f;
        const float SELU_A = 1.6732632423543772f;
        float r = (h > 0.f) ? (SELU_L * h) : (SELU_L * SELU_A * (__expf(h) - 1.f));
        out[(size_t)bi * NO + o] = r;
    }
}

extern "C" void kernel_launch(void* const* d_in, const int* in_sizes, int n_in,
                              void* d_out, int out_size, void* d_ws, size_t ws_size,
                              hipStream_t stream) {
    const float* x        = (const float*)d_in[0];
    const float* W_ap     = (const float*)d_in[1];
    const float* b_ap     = (const float*)d_in[2];
    const float* att_w    = (const float*)d_in[3];
    const float* W_pa     = (const float*)d_in[4];
    const float* b_pa     = (const float*)d_in[5];
    const float* W_po     = (const float*)d_in[6];
    const float* b_po     = (const float*)d_in[7];
    const float* bn_gamma = (const float*)d_in[8];
    const float* bn_beta  = (const float*)d_in[9];
    const float* run_mean = (const float*)d_in[10];
    const float* run_var  = (const float*)d_in[11];
    float* out = (float*)d_out;

    dim3 grid(NB * NN);   // one block per (b, i)
    dim3 block(256);      // 8 waves (wave32)
    gat_fused_kernel<<<grid, block, 0, stream>>>(
        x, W_ap, b_ap, att_w, W_pa, b_pa, W_po, b_po,
        bn_gamma, bn_beta, run_mean, run_var, out);
}
